// MultiHeadLatentAttention_755914244591
// MI455X (gfx1250) — matrix operand order repaired and checked by reading, not compiled, not run on hardware
//
#include <hip/hip_runtime.h>

// ---------------------------------------------------------------------------
// MLA block for MI455X (gfx1250, wave32, WMMA bf16 16x16x32, f32 accum).
// GEMMs on the matrix pipe; flash-attention K/V tiles staged to LDS with the
// Tensor Data Mover (double-buffered, TENSORcnt-tracked) so WMMA operand
// fetches are ds_load_b128 instead of latency-exposed global loads.
// ---------------------------------------------------------------------------

typedef __bf16 bf16;
typedef __attribute__((ext_vector_type(16))) __bf16 v16bf;
typedef __attribute__((ext_vector_type(8)))  float  v8f;
typedef __attribute__((ext_vector_type(4)))  unsigned u32x4;
typedef __attribute__((ext_vector_type(8)))  int      i32x8;
typedef __attribute__((ext_vector_type(4)))  int      i32x4;

#define B_    2
#define S_    2048
#define D_    1024
#define H_    16
#define DH_   64
#define DQL_  512
#define DKVL_ 512
#define TOPK_ 64

union FragBF { v16bf v; uint4 u[2]; };

static __device__ __forceinline__ v8f wmma_bf16(const v16bf& a, const v16bf& b, const v8f& c) {
  return __builtin_amdgcn_wmma_f32_16x16x32_bf16(false, a, false, b, (short)0, c, false, false);
}

// ---------------------------------------------------------------------------
// TDM: async tensor load (global -> LDS), 2D tile of 2-byte elements.
//   lds_addr : LDS byte offset (low 32 bits of flat shared pointer)
//   gaddr    : global address of tile start
//   td0/td1  : tensor dims (elements)   tile0/tile1 : tile dims (elements)
//   stride0  : tensor dim0 stride (elements)
// D# layout per CDNA5 ISA ch.8 (group0 128b, group1 256b; groups 2/3 zero).
// ---------------------------------------------------------------------------
static __device__ __forceinline__ void tdm_load_2d_b16(unsigned lds_addr, const void* gaddr,
                                                       unsigned td0, unsigned td1,
                                                       unsigned tile0, unsigned tile1,
                                                       unsigned long long stride0) {
  const unsigned long long ga = (unsigned long long)(uintptr_t)gaddr;
  u32x4 g0;
  g0[0] = 1u;                                    // count=1, no gather
  g0[1] = lds_addr;                              // LDS byte address
  g0[2] = (unsigned)ga;                          // global_addr[31:0]
  g0[3] = (unsigned)(ga >> 32) | (2u << 30);     // global_addr[56:32] | type=2

  i32x8 g1;
  g1[0] = (int)(1u << 16);                                   // data_size=1 -> 2 bytes
  g1[1] = (int)((td0 & 0xFFFFu) << 16);                      // tensor_dim0[15:0]
  g1[2] = (int)((td0 >> 16) | ((td1 & 0xFFFFu) << 16));      // td0[31:16] | td1[15:0]
  g1[3] = (int)((td1 >> 16) | (tile0 << 16));                // td1[31:16] | tile_dim0
  g1[4] = (int)(tile1 & 0xFFFFu);                            // tile_dim1 | tile_dim2=0
  g1[5] = (int)(unsigned)stride0;                            // stride0[31:0]
  g1[6] = (int)((unsigned)(stride0 >> 32) & 0xFFFFu);        // stride0[47:32] | stride1_lo=0
  g1[7] = 0;

  const i32x4 z4 = {0, 0, 0, 0};
#if __clang_major__ >= 23
  const i32x8 z8 = {0, 0, 0, 0, 0, 0, 0, 0};
  __builtin_amdgcn_tensor_load_to_lds(g0, g1, z4, z4, z8, 0);
#else
  __builtin_amdgcn_tensor_load_to_lds(g0, g1, z4, z4, 0);
#endif
}

// ---------------------------------------------------------------------------
// fp32 -> bf16 conversion (weights)
// ---------------------------------------------------------------------------
__global__ __launch_bounds__(256) void cvt_f32_bf16(const float* __restrict__ in,
                                                    bf16* __restrict__ out, int n) {
  for (int i = blockIdx.x * 256 + threadIdx.x; i < n; i += gridDim.x * 256)
    out[i] = (bf16)in[i];
}

// ---------------------------------------------------------------------------
// LayerNorm: one block per row of (B*S, D), emit bf16 h
// ---------------------------------------------------------------------------
__global__ __launch_bounds__(256) void layernorm_bf16(const float* __restrict__ x,
                                                      const float* __restrict__ g,
                                                      const float* __restrict__ b,
                                                      bf16* __restrict__ h) {
  __shared__ float red[256];
  const int row = blockIdx.x;
  const float* xr = x + (size_t)row * D_;
  const int t = threadIdx.x;

  float s = 0.f;
  for (int c = t; c < D_; c += 256) s += xr[c];
  red[t] = s; __syncthreads();
  for (int o = 128; o > 0; o >>= 1) { if (t < o) red[t] += red[t + o]; __syncthreads(); }
  const float mu = red[0] * (1.0f / D_);
  __syncthreads();

  float v = 0.f;
  for (int c = t; c < D_; c += 256) { float d = xr[c] - mu; v += d * d; }
  red[t] = v; __syncthreads();
  for (int o = 128; o > 0; o >>= 1) { if (t < o) red[t] += red[t + o]; __syncthreads(); }
  const float inv = rsqrtf(red[0] * (1.0f / D_) + 1e-5f);

  bf16* hr = h + (size_t)row * D_;
  for (int c = t; c < D_; c += 256)
    hr[c] = (bf16)((xr[c] - mu) * inv * g[c] + b[c]);
}

// ---------------------------------------------------------------------------
// WMMA GEMM: C[m][n] = sum_k A[m][k] * W[n][k]   (A MxK, W NxK, both bf16 K-contig)
// Block tile 128x128, 8 waves, wave tile 32x64 (2x4 WMMAs), K-step 32.
// EPI: 0 = bf16 row-major, 1 = q/k layout [b][h][s][dh], 2 = v^T [b][h][dh][s],
//      3 = fp32 + residual
// ---------------------------------------------------------------------------
template <int EPI>
__global__ __launch_bounds__(256) void gemm_wmma(const bf16* __restrict__ A,
                                                 const bf16* __restrict__ W,
                                                 bf16* __restrict__ Cb,
                                                 float* __restrict__ Cf,
                                                 const float* __restrict__ resid,
                                                 int M, int N, int K) {
  const int tid  = threadIdx.x;
  const int lane = tid & 31;
  const int w    = tid >> 5;      // 0..7
  const int wm   = w >> 1;        // 0..3
  const int wn   = w & 1;         // 0..1
  const int l15  = lane & 15;
  const int lhi  = lane >> 4;     // 0/1

  const int bm = blockIdx.y * 128;
  const int bn = blockIdx.x * 128;
  const int arow0 = bm + wm * 32;
  const int brow0 = bn + wn * 64;

  v8f acc[2][4];
#pragma unroll
  for (int i = 0; i < 2; ++i)
#pragma unroll
    for (int j = 0; j < 4; ++j) acc[i][j] = {};

  for (int k0 = 0; k0 < K; k0 += 32) {
    FragBF a[2], b[4];
#pragma unroll
    for (int i = 0; i < 2; ++i) {
      const bf16* p = A + (size_t)(arow0 + i * 16 + l15) * K + k0 + lhi * 8;
      a[i].u[0] = *(const uint4*)p;
      a[i].u[1] = *(const uint4*)(p + 16);
    }
#pragma unroll
    for (int j = 0; j < 4; ++j) {
      const bf16* p = W + (size_t)(brow0 + j * 16 + l15) * K + k0 + lhi * 8;
      b[j].u[0] = *(const uint4*)p;
      b[j].u[1] = *(const uint4*)(p + 16);
    }
    if (k0 + 32 < K) {   // lowers to global_prefetch_b8 on gfx1250
      __builtin_prefetch(A + (size_t)(arow0 + l15) * K + k0 + 32, 0, 0);
      __builtin_prefetch(W + (size_t)(brow0 + l15) * K + k0 + 32, 0, 0);
    }
#pragma unroll
    for (int i = 0; i < 2; ++i)
#pragma unroll
      for (int j = 0; j < 4; ++j)
        acc[i][j] = wmma_bf16(a[i].v, b[j].v, acc[i][j]);
  }

#pragma unroll
  for (int i = 0; i < 2; ++i)
#pragma unroll
    for (int j = 0; j < 4; ++j)
#pragma unroll
      for (int e = 0; e < 8; ++e) {
        const int m = arow0 + i * 16 + lhi * 8 + e;
        const int n = brow0 + j * 16 + l15;
        const float val = acc[i][j][e];
        if constexpr (EPI == 0) {
          Cb[(size_t)m * N + n] = (bf16)val;
        } else if constexpr (EPI == 1) {     // [b][h][s][dh]
          const int bb = m >> 11, s = m & (S_ - 1);
          const int hh = n >> 6, dh = n & 63;
          Cb[(((size_t)(bb * H_ + hh)) * S_ + s) * DH_ + dh] = (bf16)val;
        } else if constexpr (EPI == 2) {     // [b][h][dh][s]
          const int bb = m >> 11, s = m & (S_ - 1);
          const int hh = n >> 6, dh = n & 63;
          Cb[(((size_t)(bb * H_ + hh)) * DH_ + dh) * S_ + s] = (bf16)val;
        } else {                             // fp32 + residual
          Cf[(size_t)m * N + n] = val + resid[(size_t)m * N + n];
        }
      }
}

// ---------------------------------------------------------------------------
// RoPE over [B*H*S][64] bf16, in place. cos/sin index map matches reference:
// freq index f(i) = (2i) mod 32, inv_freq[j] = 10000^(-2j/64).
// ---------------------------------------------------------------------------
__global__ __launch_bounds__(256) void rope_kernel(bf16* __restrict__ t) {
  const int row = blockIdx.x * 8 + (threadIdx.x >> 5);  // over B*H*S rows
  const int p   = threadIdx.x & 31;                     // pair index
  const int s   = row & (S_ - 1);
  bf16* r = t + (size_t)row * DH_;
  const float e = (float)r[2 * p];
  const float o = (float)r[2 * p + 1];
  const int   f = (2 * p) & 31;
  const float invf = powf(10000.0f, -((float)(2 * f)) / 64.0f);
  const float ang  = (float)s * invf;
  const float c = cosf(ang), sn = sinf(ang);
  r[2 * p]     = (bf16)(e * c - o * sn);
  r[2 * p + 1] = (bf16)(e * sn + o * c);
}

// ---------------------------------------------------------------------------
// Flash attention + streaming top-64.
// Grid: (S/64, B*H). Block: 256 threads = 8 waves.
// Q tile (64x64) in registers as A-fragments.  K/V tiles double-buffered in
// LDS via TDM (wave 0 issues tensor_load_to_lds for tile kt+1 while all
// waves compute on tile kt; s_wait_tensorcnt + barrier hand-off).
// ---------------------------------------------------------------------------
__global__ __launch_bounds__(256) void flash_topk(const bf16* __restrict__ Q,
                                                  const bf16* __restrict__ Kk,
                                                  const bf16* __restrict__ Vt,
                                                  bf16* __restrict__ Oattn,
                                                  long long* __restrict__ IdxOut) {
  __shared__ bf16  sK[2][64 * 64];   // K tile: [key][dh]
  __shared__ bf16  sV[2][64 * 64];   // V tile: [dh][key]
  __shared__ float sS[64 * 64];      // raw scores (f32)
  __shared__ bf16  sP[64 * 64];      // probabilities (bf16)
  __shared__ float sM[64], sL[64], sC[64];
  __shared__ float tkv[64 * 64];     // topk values
  __shared__ int   tki[64 * 64];     // topk indices

  const int bh  = blockIdx.y;                 // b*H + h
  const int qb  = blockIdx.x;                 // query block
  const int tid = threadIdx.x;
  const int lane = tid & 31;
  const int w    = tid >> 5;
  const int msub  = w >> 1;                   // 0..3
  const int npair = w & 1;                    // 0..1
  const int l15 = lane & 15, lhi = lane >> 4;

  const bf16* Qbh = Q  + (size_t)bh * S_ * DH_;
  const bf16* Kbh = Kk + (size_t)bh * S_ * DH_;
  const bf16* Vbh = Vt + (size_t)bh * DH_ * S_;

  // Q fragments (A operand), reused for all key tiles
  FragBF qf[2];
  {
    const bf16* p = Qbh + (size_t)(qb * 64 + msub * 16 + l15) * DH_;
#pragma unroll
    for (int kk = 0; kk < 2; ++kk) {
      qf[kk].u[0] = *(const uint4*)(p + kk * 32 + lhi * 8);
      qf[kk].u[1] = *(const uint4*)(p + kk * 32 + lhi * 8 + 16);
    }
  }

  v8f oacc[2]; oacc[0] = {}; oacc[1] = {};
  if (tid < 64) { sM[tid] = -1e30f; sL[tid] = 0.f; }
  float minv = 1e30f; int minpos = 0;

  const int NT = S_ / 64;

  // preload tile 0 (wave 0 drives the TDM; TENSORcnt is per-wave)
  if (w == 0) {
    tdm_load_2d_b16((unsigned)(uintptr_t)&sK[0][0], Kbh, /*td0*/64, /*td1*/64,
                    /*tile0*/64, /*tile1*/64, /*stride0*/64);
    tdm_load_2d_b16((unsigned)(uintptr_t)&sV[0][0], Vbh, /*td0*/S_, /*td1*/64,
                    /*tile0*/64, /*tile1*/64, /*stride0*/S_);
    __builtin_amdgcn_s_wait_tensorcnt(0);
  }
  __syncthreads();

  const float scale = 0.125f;  // 1/sqrt(64)

  for (int kt = 0; kt < NT; ++kt) {
    const int cur = kt & 1, nxt = cur ^ 1;

    // ---- prefetch next K/V tiles via TDM while we compute on `cur` ----
    if (w == 0 && (kt + 1) < NT) {
      tdm_load_2d_b16((unsigned)(uintptr_t)&sK[nxt][0],
                      Kbh + (size_t)(kt + 1) * 64 * DH_, 64, 64, 64, 64, 64);
      tdm_load_2d_b16((unsigned)(uintptr_t)&sV[nxt][0],
                      Vbh + (size_t)(kt + 1) * 64, S_, 64, 64, 64, S_);
    }

    // ---- 1. score tile S = Q K^T (K frags from LDS) ----
    {
      FragBF kf[2][2];
#pragma unroll
      for (int jj = 0; jj < 2; ++jj) {
        const int nsub = npair * 2 + jj;
        const bf16* p = &sK[cur][(nsub * 16 + l15) * 64 + lhi * 8];
        kf[jj][0].u[0] = *(const uint4*)(p);
        kf[jj][0].u[1] = *(const uint4*)(p + 16);
        kf[jj][1].u[0] = *(const uint4*)(p + 32);
        kf[jj][1].u[1] = *(const uint4*)(p + 48);
      }
#pragma unroll
      for (int jj = 0; jj < 2; ++jj) {
        const int nsub = npair * 2 + jj;
        v8f sc = {};
        sc = wmma_bf16(qf[0].v, kf[jj][0].v, sc);
        sc = wmma_bf16(qf[1].v, kf[jj][1].v, sc);
#pragma unroll
        for (int e = 0; e < 8; ++e)
          sS[(msub * 16 + lhi * 8 + e) * 64 + nsub * 16 + l15] = sc[e] * scale;
      }
    }
    __syncthreads();

    // ---- 2. online softmax + streaming top-64 (threads 0..63, one row each) ----
    if (tid < 64) {
      const float mold = sM[tid];
      float tmax = -1e30f;
      for (int j = 0; j < 64; ++j) tmax = fmaxf(tmax, sS[tid * 64 + j]);
      const float mnew = fmaxf(mold, tmax);
      const float corr = __expf(mold - mnew);
      float tsum = 0.f;
      for (int j = 0; j < 64; ++j) {
        const float sv = sS[tid * 64 + j];
        const float pv = __expf(sv - mnew);
        tsum += pv;
        sP[tid * 64 + j] = (bf16)pv;
        if (kt == 0) {
          tkv[tid * 64 + j] = sv; tki[tid * 64 + j] = j;
        } else if (sv > minv) {
          tkv[tid * 64 + minpos] = sv; tki[tid * 64 + minpos] = kt * 64 + j;
          minv = 1e30f;
          for (int q2 = 0; q2 < 64; ++q2) {
            const float vv = tkv[tid * 64 + q2];
            if (vv < minv) { minv = vv; minpos = q2; }
          }
        }
      }
      if (kt == 0) {
        minv = 1e30f;
        for (int q2 = 0; q2 < 64; ++q2) {
          const float vv = tkv[tid * 64 + q2];
          if (vv < minv) { minv = vv; minpos = q2; }
        }
      }
      sL[tid] = sL[tid] * corr + tsum;
      sM[tid] = mnew;
      sC[tid] = corr;
    }
    __syncthreads();

    // ---- 3. rescale accumulators + O += P V (P and V frags from LDS) ----
    FragBF pf[2];
#pragma unroll
    for (int kk = 0; kk < 2; ++kk) {
      const bf16* p = sP + (msub * 16 + l15) * 64 + kk * 32 + lhi * 8;
      pf[kk].u[0] = *(const uint4*)p;
      pf[kk].u[1] = *(const uint4*)(p + 16);
    }
#pragma unroll
    for (int jj = 0; jj < 2; ++jj) {
#pragma unroll
      for (int e = 0; e < 8; ++e)
        oacc[jj][e] *= sC[msub * 16 + lhi * 8 + e];
      const int nsub = npair * 2 + jj;
      FragBF vf[2];
#pragma unroll
      for (int kk = 0; kk < 2; ++kk) {
        const bf16* p = &sV[cur][(nsub * 16 + l15) * 64 + kk * 32 + lhi * 8];
        vf[kk].u[0] = *(const uint4*)p;
        vf[kk].u[1] = *(const uint4*)(p + 16);
      }
      oacc[jj] = wmma_bf16(pf[0].v, vf[0].v, oacc[jj]);
      oacc[jj] = wmma_bf16(pf[1].v, vf[1].v, oacc[jj]);
    }

    // next buffer must be resident before anyone reads it next iteration
    if (w == 0) __builtin_amdgcn_s_wait_tensorcnt(0);
    __syncthreads();
  }

  // ---- finalize: divide by l, store O (as [b][s][h*64+dh] bf16) ----
  if (tid < 64) sC[tid] = 1.0f / sL[tid];
  __syncthreads();

  const int bb = bh >> 4, hh = bh & 15;
#pragma unroll
  for (int jj = 0; jj < 2; ++jj) {
    const int nsub = npair * 2 + jj;
#pragma unroll
    for (int e = 0; e < 8; ++e) {
      const int m    = msub * 16 + lhi * 8 + e;
      const int srow = qb * 64 + m;
      const int dh   = nsub * 16 + l15;
      const float val = oacc[jj][e] * sC[m];
      Oattn[((size_t)(bb * S_ + srow)) * (H_ * DH_) + hh * DH_ + dh] = (bf16)val;
    }
  }

  // ---- top-k: sort descending (selection sort) and emit int64 indices ----
  if (tid < 64) {
    const int srow = qb * 64 + tid;
    long long* dst = IdxOut + ((size_t)bh * S_ + srow) * TOPK_;
    float* vv = tkv + tid * 64;
    int*   ii = tki + tid * 64;
    for (int r = 0; r < TOPK_; ++r) {
      float best = -1e38f; int bp = r;
      for (int j = r; j < 64; ++j)
        if (vv[j] > best) { best = vv[j]; bp = j; }
      const float tv = vv[r]; vv[r] = vv[bp]; vv[bp] = tv;
      const int   ti = ii[r]; ii[r] = ii[bp]; ii[bp] = ti;
      dst[r] = (long long)ii[r];
    }
  }
}

// ---------------------------------------------------------------------------
// Launcher
// ---------------------------------------------------------------------------
extern "C" void kernel_launch(void* const* d_in, const int* in_sizes, int n_in,
                              void* d_out, int out_size, void* d_ws, size_t ws_size,
                              hipStream_t stream) {
  const float* x    = (const float*)d_in[0];
  const float* wdq  = (const float*)d_in[1];
  const float* wdkv = (const float*)d_in[2];
  const float* wupq = (const float*)d_in[3];
  const float* wupk = (const float*)d_in[4];
  const float* wupv = (const float*)d_in[5];
  const float* wout = (const float*)d_in[6];
  const float* lns  = (const float*)d_in[7];
  const float* lnb  = (const float*)d_in[8];

  char* ws = (char*)d_ws;
  size_t off = 0;
  auto take = [&](size_t elems) -> bf16* {
    bf16* p = (bf16*)(ws + off);
    off += ((elems * sizeof(bf16) + 255) / 256) * 256;
    return p;
  };

  const size_t MROWS = (size_t)B_ * S_;        // 4096
  bf16* hbuf  = take(MROWS * D_);
  bf16* wdqb  = take((size_t)DQL_ * D_);
  bf16* wdkvb = take((size_t)DKVL_ * D_);
  bf16* wupqb = take((size_t)H_ * DH_ * DQL_);
  bf16* wupkb = take((size_t)H_ * DH_ * DKVL_);
  bf16* wupvb = take((size_t)H_ * DH_ * DKVL_);
  bf16* woutb = take((size_t)D_ * H_ * DH_);
  bf16* cq    = take(MROWS * DQL_);
  bf16* ckv   = take(MROWS * DKVL_);
  bf16* qb    = take((size_t)B_ * H_ * S_ * DH_);
  bf16* kb    = take((size_t)B_ * H_ * S_ * DH_);
  bf16* vtb   = take((size_t)B_ * H_ * DH_ * S_);
  bf16* attnb = take(MROWS * (size_t)H_ * DH_);

  float*     out_f   = (float*)d_out;
  long long* out_idx = (long long*)(out_f + MROWS * D_);

  // 1. weights -> bf16
  auto cvt = [&](const float* src, bf16* dst, int n) {
    cvt_f32_bf16<<<(n + 1023) / 1024, 256, 0, stream>>>(src, dst, n);
  };
  cvt(wdq,  wdqb,  DQL_ * D_);
  cvt(wdkv, wdkvb, DKVL_ * D_);
  cvt(wupq, wupqb, H_ * DH_ * DQL_);
  cvt(wupk, wupkb, H_ * DH_ * DKVL_);
  cvt(wupv, wupvb, H_ * DH_ * DKVL_);
  cvt(wout, woutb, D_ * H_ * DH_);

  // 2. LayerNorm -> h (bf16)
  layernorm_bf16<<<MROWS, 256, 0, stream>>>(x, lns, lnb, hbuf);

  // 3. down projections: c_q = h W_dq^T, c_kv = h W_dkv^T
  gemm_wmma<0><<<dim3(DQL_ / 128, MROWS / 128), 256, 0, stream>>>(
      hbuf, wdqb, cq, nullptr, nullptr, (int)MROWS, DQL_, D_);
  gemm_wmma<0><<<dim3(DKVL_ / 128, MROWS / 128), 256, 0, stream>>>(
      hbuf, wdkvb, ckv, nullptr, nullptr, (int)MROWS, DKVL_, D_);

  // 4. up projections with layout-transforming epilogues
  gemm_wmma<1><<<dim3((H_ * DH_) / 128, MROWS / 128), 256, 0, stream>>>(
      cq, wupqb, qb, nullptr, nullptr, (int)MROWS, H_ * DH_, DQL_);
  gemm_wmma<1><<<dim3((H_ * DH_) / 128, MROWS / 128), 256, 0, stream>>>(
      ckv, wupkb, kb, nullptr, nullptr, (int)MROWS, H_ * DH_, DKVL_);
  gemm_wmma<2><<<dim3((H_ * DH_) / 128, MROWS / 128), 256, 0, stream>>>(
      ckv, wupvb, vtb, nullptr, nullptr, (int)MROWS, H_ * DH_, DKVL_);

  // 5. RoPE (in place, q and k)
  rope_kernel<<<(B_ * H_ * S_) / 8, 256, 0, stream>>>(qb);
  rope_kernel<<<(B_ * H_ * S_) / 8, 256, 0, stream>>>(kb);

  // 6. flash attention + top-64 indices (TDM-staged K/V tiles)
  flash_topk<<<dim3(S_ / 64, B_ * H_), 256, 0, stream>>>(qb, kb, vtb, attnb, out_idx);

  // 7. output projection + residual (fp32 out)
  gemm_wmma<3><<<dim3(D_ / 128, MROWS / 128), 256, 0, stream>>>(
      attnb, woutb, nullptr, out_f, x, (int)MROWS, D_, H_ * DH_);
}
